// MyModel_22917945492067
// MI455X (gfx1250) — matrix operand order repaired and checked by reading, not compiled, not run on hardware
//
#include <hip/hip_runtime.h>
#include <hip/hip_bf16.h>
#include <math.h>

typedef __attribute__((ext_vector_type(16))) __bf16 v16bf;
typedef __attribute__((ext_vector_type(8)))  float  v8f;
typedef __attribute__((ext_vector_type(4)))  int    v4i;
typedef __attribute__((address_space(1)))    v4i    gv4i;   // global 16B vector
typedef __attribute__((address_space(3)))    v4i    lv4i;   // LDS 16B vector

union BFrag { uint4 q[2]; v16bf v; };

#if defined(__has_builtin)
#if __has_builtin(__builtin_amdgcn_global_load_async_to_lds_b128)
#define HAS_ASYNC_LDS 1
#endif
#endif
#ifndef HAS_ASYNC_LDS
#define HAS_ASYNC_LDS 0
#warning "gfx1250 async-to-LDS builtin not available; using synchronous staging fallback"
#endif

__device__ __forceinline__ void wait_async_zero() {
#if defined(__has_builtin)
#if __has_builtin(__builtin_amdgcn_s_wait_asynccnt)
    __builtin_amdgcn_s_wait_asynccnt(0);
#else
    asm volatile("s_wait_asynccnt 0x0" ::: "memory");
#endif
#else
    asm volatile("s_wait_asynccnt 0x0" ::: "memory");
#endif
}

// ---------------------------------------------------------------------------
// Kernel 0: zero the split-K accumulator (256x512 f32)
// ---------------------------------------------------------------------------
__global__ __launch_bounds__(256) void zero_acc(float* __restrict__ acc, int n) {
    int i = blockIdx.x * 256 + threadIdx.x;
    if (i < n) acc[i] = 0.0f;
}

// ---------------------------------------------------------------------------
// Kernel 1: scoring GEMM  acc[256,512] += xf[256,230400] @ w1.T
//   grid = (8 N-tiles, 60 K-chunks), block = 256 (8 waves)
//   A slab: fp32 async-DMA'd to LDS (double buffered), fp32->bf16 at
//           fragment read (co-executes with XDL WMMA).
//   B slab: fp32->bf16 at staging (read 8x per wave from LDS, cvt once).
//   Core:   v_wmma_f32_16x16x32_bf16; split-K merge: global_atomic_add_f32.
// ---------------------------------------------------------------------------
#define SG_M      256
#define SG_NT     64
#define SG_KS     32          // K per stage (one WMMA K-step)
#define SG_STAGES 120         // KCHUNK = 3840
#define SG_LDA    36          // padded fp32 row stride for A (bank-friendly)
#define SG_LDB    40          // padded bf16 row stride for B
#define SG_K      230400

__global__ __launch_bounds__(256) void score_gemm(const float* __restrict__ x,
                                                  const float* __restrict__ w1,
                                                  float* __restrict__ acc) {
    __shared__ __attribute__((aligned(16))) float  As[2][SG_M * SG_LDA];   // 73728 B
    __shared__ __attribute__((aligned(16))) __bf16 Bs[2][SG_NT * SG_LDB];  // 10240 B

    const int nTile  = blockIdx.x * SG_NT;
    const int kChunk = blockIdx.y * (SG_KS * SG_STAGES);
    const int tid   = threadIdx.x;
    const int wave  = tid >> 5;
    const int lane  = tid & 31;
    const int l15   = lane & 15;
    const int lHi   = lane >> 4;            // 0 or 1 (half-wave)
    const int mWave = wave * 32;            // each wave owns 32 rows of M
    const int kOffA = lHi << 3;             // A frag K base: 0 or 8
    const int kOffB = lHi << 3;             // B frag K base: 0 or 8

    // ---- stage-copy helpers (lambdas keep addressing uniform) ----
    auto copyA = [&](int s, int buf) {      // 256 rows x 32 K fp32 = 2048 x 16B
        const int kBase = kChunk + s * SG_KS;
#if HAS_ASYNC_LDS
        const unsigned long long gx = (unsigned long long)(uintptr_t)x;
        const unsigned ldsBase = (unsigned)(uintptr_t)(&As[buf][0]);
        #pragma unroll
        for (int i = 0; i < 8; ++i) {
            const int q   = tid + i * 256;
            const int row = q >> 3;         // 8 float4 per row
            const int c4  = q & 7;
            const unsigned long long ga =
                gx + (((size_t)row * SG_K + kBase + c4 * 4) << 2);
            const unsigned la = ldsBase + (unsigned)(row * SG_LDA + c4 * 4) * 4u;
            __builtin_amdgcn_global_load_async_to_lds_b128(
                (gv4i*)(uintptr_t)ga, (lv4i*)(uintptr_t)la, 0, 0);
        }
#else
        #pragma unroll
        for (int i = 0; i < 8; ++i) {
            const int q   = tid + i * 256;
            const int row = q >> 3;
            const int c4  = q & 7;
            const float4 v = *(const float4*)(x + (size_t)row * SG_K + kBase + c4 * 4);
            *(float4*)&As[buf][row * SG_LDA + c4 * 4] = v;
        }
#endif
    };
    auto copyB = [&](int s, int buf) {      // 64 rows x 32 K, cvt to bf16
        const int kBase = kChunk + s * SG_KS;
        #pragma unroll
        for (int i = 0; i < 2; ++i) {
            const int q  = tid + i * 256;
            const int n  = q >> 3;
            const int c4 = q & 7;
            const float4 v = *(const float4*)(w1 + (size_t)(nTile + n) * SG_K + kBase + c4 * 4);
            __bf16* dst = &Bs[buf][n * SG_LDB + c4 * 4];
            dst[0] = (__bf16)v.x; dst[1] = (__bf16)v.y;
            dst[2] = (__bf16)v.z; dst[3] = (__bf16)v.w;
        }
    };

    v8f c[2][4];
    const v8f vzero = {0.f,0.f,0.f,0.f,0.f,0.f,0.f,0.f};
    #pragma unroll
    for (int i = 0; i < 2; ++i)
        #pragma unroll
        for (int j = 0; j < 4; ++j) c[i][j] = vzero;

    // ---- prologue: stage 0 into buffer 0 ----
    copyA(0, 0);
    copyB(0, 0);
#if HAS_ASYNC_LDS
    wait_async_zero();
#endif
    __syncthreads();

    for (int s = 0; s < SG_STAGES; ++s) {
        const int buf  = s & 1;
        const int nbuf = buf ^ 1;
        if (s + 1 < SG_STAGES) {            // overlap next-stage DMA with compute
            copyA(s + 1, nbuf);
            copyB(s + 1, nbuf);
        }

        // ---- A fragments: read fp32 from LDS, cvt -> bf16 (ISA 16-bit A layout) ----
        v16bf a[2], b[4];
        #pragma unroll
        for (int i = 0; i < 2; ++i) {
            const int row = mWave + i * 16 + l15;
            const float4* p0 = (const float4*)&As[buf][row * SG_LDA + kOffA];
            const float4* p1 = (const float4*)&As[buf][row * SG_LDA + kOffA + 16];
            const float4 f0 = p0[0], f1 = p0[1];   // K kOffA..kOffA+7
            const float4 f2 = p1[0], f3 = p1[1];   // K kOffA+16..kOffA+23
            v16bf av;
            av[0]=(__bf16)f0.x; av[1]=(__bf16)f0.y; av[2]=(__bf16)f0.z; av[3]=(__bf16)f0.w;
            av[4]=(__bf16)f1.x; av[5]=(__bf16)f1.y; av[6]=(__bf16)f1.z; av[7]=(__bf16)f1.w;
            av[8]=(__bf16)f2.x; av[9]=(__bf16)f2.y; av[10]=(__bf16)f2.z; av[11]=(__bf16)f2.w;
            av[12]=(__bf16)f3.x; av[13]=(__bf16)f3.y; av[14]=(__bf16)f3.z; av[15]=(__bf16)f3.w;
            a[i] = av;
        }
        // ---- B fragments: lane = N, K kOffB..+7 and kOffB+16..+23 per half-wave ----
        #pragma unroll
        for (int j = 0; j < 4; ++j) {
            const int n = j * 16 + l15;
            BFrag u;
            u.q[0] = *(const uint4*)&Bs[buf][n * SG_LDB + kOffB];
            u.q[1] = *(const uint4*)&Bs[buf][n * SG_LDB + kOffB + 16];
            b[j] = u.v;
        }
        #pragma unroll
        for (int i = 0; i < 2; ++i)
            #pragma unroll
            for (int j = 0; j < 4; ++j)
                c[i][j] = __builtin_amdgcn_wmma_f32_16x16x32_bf16(
                    false, a[i], false, b[j], (short)0, c[i][j], false, false);

#if HAS_ASYNC_LDS
        wait_async_zero();                  // next-stage A landed in LDS
#endif
        __syncthreads();                    // B visible; all reads of nbuf done
    }

    // ---- split-K merge: C layout (VGPR r, lane): M = base + lHi*8 + r, N = l15 ----
    #pragma unroll
    for (int i = 0; i < 2; ++i) {
        const int rowBase = mWave + i * 16 + lHi * 8;
        #pragma unroll
        for (int j = 0; j < 4; ++j) {
            const int col = nTile + j * 16 + l15;
            #pragma unroll
            for (int r = 0; r < 8; ++r)
                atomicAdd(&acc[(size_t)(rowBase + r) * 512 + col], c[i][j][r]);
        }
    }
}

// ---------------------------------------------------------------------------
// Kernel 2: bias + ReLU + dot(w2) -> scores[32], then top-5 (descending).
// ---------------------------------------------------------------------------
__global__ __launch_bounds__(256) void score_topk(const float* __restrict__ acc,
                                                  const float* __restrict__ b1,
                                                  const float* __restrict__ w2,
                                                  const float* __restrict__ b2,
                                                  int* __restrict__ idx) {
    __shared__ float sc[32];
    const int b    = blockIdx.x;
    const int wave = threadIdx.x >> 5;
    const int lane = threadIdx.x & 31;

    for (int f = wave; f < 32; f += 8) {
        const float* row = acc + (size_t)(b * 32 + f) * 512;
        float sum = 0.f;
        #pragma unroll 4
        for (int n = lane; n < 512; n += 32) {
            float h = row[n] + b1[n];
            h = h > 0.f ? h : 0.f;
            sum += h * w2[n];
        }
        #pragma unroll
        for (int off = 16; off > 0; off >>= 1) sum += __shfl_xor(sum, off, 32);
        if (lane == 0) sc[f] = sum + b2[0];
    }
    __syncthreads();

    if (threadIdx.x == 0) {
        float v[32];
        #pragma unroll
        for (int i = 0; i < 32; ++i) v[i] = sc[i];
        for (int k = 0; k < 5; ++k) {           // descending, first-index on ties
            int best = 0; float bv = v[0];
            for (int i = 1; i < 32; ++i) if (v[i] > bv) { bv = v[i]; best = i; }
            idx[b * 5 + k] = best;
            v[best] = -3.402823e38f;
        }
    }
}

// ---------------------------------------------------------------------------
// Kernel 3: gathered conv stem (3->16, k3, s2, p1) + ReLU + GAP -> gap[40,16]
// ---------------------------------------------------------------------------
__global__ __launch_bounds__(256) void conv_gap(const float* __restrict__ x,
                                                const int* __restrict__ idx,
                                                const float* __restrict__ conv_w,
                                                float* __restrict__ gap) {
    __shared__ float csum[16];
    __shared__ float wsm[432];           // 16*3*3*3
    const int j = blockIdx.x;            // frame slot 0..39
    const int b = j / 5;
    const int fsel = idx[j];
    const float* img = x + ((size_t)b * 32 + fsel) * 230400;   // [3,240,320]

    if (threadIdx.x < 16) csum[threadIdx.x] = 0.f;
    for (int i = threadIdx.x; i < 432; i += 256) wsm[i] = conv_w[i];
    __syncthreads();

    float accl[16];
    #pragma unroll
    for (int o = 0; o < 16; ++o) accl[o] = 0.f;

    for (int p = threadIdx.x; p < 120 * 160; p += 256) {
        const int oy = p / 160, ox = p - oy * 160;
        const int iy0 = oy * 2 - 1, ix0 = ox * 2 - 1;
        float in[3][3][3];
        #pragma unroll
        for (int ci = 0; ci < 3; ++ci)
            #pragma unroll
            for (int ky = 0; ky < 3; ++ky)
                #pragma unroll
                for (int kx = 0; kx < 3; ++kx) {
                    const int iy = iy0 + ky, ix = ix0 + kx;
                    in[ci][ky][kx] = (iy >= 0 && iy < 240 && ix >= 0 && ix < 320)
                                   ? img[ci * 76800 + iy * 320 + ix] : 0.f;
                }
        #pragma unroll
        for (int o = 0; o < 16; ++o) {
            float s = 0.f;
            #pragma unroll
            for (int ci = 0; ci < 3; ++ci)
                #pragma unroll
                for (int ky = 0; ky < 3; ++ky)
                    #pragma unroll
                    for (int kx = 0; kx < 3; ++kx)
                        s += in[ci][ky][kx] * wsm[((o * 3 + ci) * 3 + ky) * 3 + kx];
            accl[o] += s > 0.f ? s : 0.f;      // ReLU before GAP
        }
    }
    #pragma unroll
    for (int o = 0; o < 16; ++o) atomicAdd(&csum[o], accl[o]);  // ds_add_f32
    __syncthreads();
    if (threadIdx.x < 16)
        gap[j * 16 + threadIdx.x] = csum[threadIdx.x] * (1.f / 19200.f);
}

// ---------------------------------------------------------------------------
// Kernel 4: gap[40,16] -> img_w(1280x16)+b -> fc1(101x1280)+b -> feat101[40,101]
// ---------------------------------------------------------------------------
__global__ __launch_bounds__(128) void fc_head(const float* __restrict__ gap,
                                               const float* __restrict__ img_w,
                                               const float* __restrict__ img_b,
                                               const float* __restrict__ fc1_w,
                                               const float* __restrict__ fc1_b,
                                               float* __restrict__ feat101) {
    __shared__ float e[1280];
    __shared__ float g[16];
    const int j = blockIdx.x;
    if (threadIdx.x < 16) g[threadIdx.x] = gap[j * 16 + threadIdx.x];
    __syncthreads();
    for (int n = threadIdx.x; n < 1280; n += 128) {
        float s = img_b[n];
        #pragma unroll
        for (int k = 0; k < 16; ++k) s += g[k] * img_w[n * 16 + k];
        e[n] = s;
    }
    __syncthreads();
    for (int o = threadIdx.x; o < 101; o += 128) {
        float s = fc1_b[o];
        #pragma unroll 8
        for (int k = 0; k < 1280; ++k) s += e[k] * fc1_w[o * 1280 + k];
        feat101[j * 101 + o] = s;
    }
}

// ---------------------------------------------------------------------------
// Kernel 5: LSTM(101->128) over 5 steps, mean-pool, fc2(128->101) -> out[8,101]
// ---------------------------------------------------------------------------
__global__ __launch_bounds__(128) void lstm_head(const float* __restrict__ feat101,
                                                 const float* __restrict__ w_ih,
                                                 const float* __restrict__ w_hh,
                                                 const float* __restrict__ b_ih,
                                                 const float* __restrict__ b_hh,
                                                 const float* __restrict__ fc2_w,
                                                 const float* __restrict__ fc2_b,
                                                 float* __restrict__ out) {
    __shared__ float h[128];
    __shared__ float xs[101];
    __shared__ float pooled[128];
    const int b  = blockIdx.x;
    const int t0 = threadIdx.x;
    float cstate = 0.f, hsum = 0.f;
    h[t0] = 0.f;
    __syncthreads();

    for (int t = 0; t < 5; ++t) {
        if (t0 < 101) xs[t0] = feat101[(b * 5 + t) * 101 + t0];
        __syncthreads();
        float gv[4];                      // torch gate order: i, f, g, o
        #pragma unroll
        for (int gi = 0; gi < 4; ++gi) {
            const int r = gi * 128 + t0;
            float s = b_ih[r] + b_hh[r];
            for (int k = 0; k < 101; ++k) s += xs[k] * w_ih[r * 101 + k];
            for (int k = 0; k < 128; ++k) s += h[k] * w_hh[r * 128 + k];
            gv[gi] = s;
        }
        __syncthreads();                  // all reads of h done
        const float ig = 1.f / (1.f + __expf(-gv[0]));
        const float fg = 1.f / (1.f + __expf(-gv[1]));
        const float gg = tanhf(gv[2]);
        const float og = 1.f / (1.f + __expf(-gv[3]));
        cstate = fg * cstate + ig * gg;
        const float hn = og * tanhf(cstate);
        h[t0] = hn;
        hsum += hn;
    }
    pooled[t0] = hsum * 0.2f;
    __syncthreads();
    if (t0 < 101) {
        float s = fc2_b[t0];
        #pragma unroll 8
        for (int k = 0; k < 128; ++k) s += pooled[k] * fc2_w[t0 * 128 + k];
        out[b * 101 + t0] = s;
    }
}

// ---------------------------------------------------------------------------
extern "C" void kernel_launch(void* const* d_in, const int* in_sizes, int n_in,
                              void* d_out, int out_size, void* d_ws, size_t ws_size,
                              hipStream_t stream) {
    const float* x      = (const float*)d_in[0];
    const float* mlp_w1 = (const float*)d_in[1];
    const float* mlp_b1 = (const float*)d_in[2];
    const float* mlp_w2 = (const float*)d_in[3];
    const float* mlp_b2 = (const float*)d_in[4];
    const float* conv_w = (const float*)d_in[5];
    const float* img_w  = (const float*)d_in[6];
    const float* img_b  = (const float*)d_in[7];
    const float* fc1_w  = (const float*)d_in[8];
    const float* fc1_b  = (const float*)d_in[9];
    const float* w_ih   = (const float*)d_in[10];
    const float* w_hh   = (const float*)d_in[11];
    const float* b_ih   = (const float*)d_in[12];
    const float* b_hh   = (const float*)d_in[13];
    const float* fc2_w  = (const float*)d_in[14];
    const float* fc2_b  = (const float*)d_in[15];

    float* ws      = (float*)d_ws;
    float* acc     = ws;                       // 256*512      = 131072 f32
    int*   idx     = (int*)(ws + 131072);      // 40 ints (pad to 64)
    float* gap     = ws + 131072 + 64;         // 40*16        = 640 f32
    float* feat101 = gap + 640;                // 40*101       = 4040 f32

    zero_acc<<<512, 256, 0, stream>>>(acc, 256 * 512);
    score_gemm<<<dim3(8, 60), 256, 0, stream>>>(x, mlp_w1, acc);
    score_topk<<<8, 256, 0, stream>>>(acc, mlp_b1, mlp_w2, mlp_b2, idx);
    conv_gap<<<40, 256, 0, stream>>>(x, idx, conv_w, gap);
    fc_head<<<40, 128, 0, stream>>>(gap, img_w, img_b, fc1_w, fc1_b, feat101);
    lstm_head<<<8, 128, 0, stream>>>(feat101, w_ih, w_hh, b_ih, b_hh,
                                     fc2_w, fc2_b, (float*)d_out);
}
